// ROIViTExtractor_19911468384295
// MI455X (gfx1250) — matrix operand order repaired and checked by reading, not compiled, not run on hardware
//
#include <hip/hip_runtime.h>
#include <stdint.h>
#include <stddef.h>

// ---------------- model dimensions ----------------
#define D_MODEL 768
#define L_TOK   50
#define NROI    64
#define NHEAD   12
#define HDIM    64
#define MLPD    3072
#define NLAYER  12
#define NPATCH  49
#define ROWS    (NROI * L_TOK)   // 3200

// ---------------- vector types ----------------
typedef __attribute__((ext_vector_type(16))) __bf16       bf16x16;
typedef __attribute__((ext_vector_type(8)))  float        f32x8;
typedef __attribute__((ext_vector_type(4)))  float        f32x4;
typedef __attribute__((ext_vector_type(4)))  unsigned int u32x4;
typedef int v4i_vs __attribute__((vector_size(16)));      // matches builtin pointee

union FragAB { u32x4 u[2]; bf16x16 v; };   // 32 B = one WMMA A/B fragment

__device__ __forceinline__ uint16_t f2bf(float f) {
  uint32_t u = __float_as_uint(f);
  u += 0x7FFFu + ((u >> 16) & 1u);     // round-to-nearest-even
  return (uint16_t)(u >> 16);
}

__device__ __forceinline__ f32x8 wmma_bf16(const FragAB& a, const FragAB& b, f32x8 c) {
  return __builtin_amdgcn_wmma_f32_16x16x32_bf16(false, a.v, false, b.v,
                                                 (short)0, c, false, false);
}

// ---------------- async global->LDS copy (CDNA5), guarded ----------------
#if __has_builtin(__builtin_amdgcn_global_load_async_to_lds_b128)
#define HAVE_ASYNC_LDS 1
#else
#define HAVE_ASYNC_LDS 0
#endif

__device__ __forceinline__ void cp16(const uint16_t* g, uint16_t* l) {
#if HAVE_ASYNC_LDS
  __builtin_amdgcn_global_load_async_to_lds_b128(
      (__attribute__((address_space(1))) v4i_vs*)(uintptr_t)g,
      (__attribute__((address_space(3))) v4i_vs*)(uint32_t)(uintptr_t)l,
      0, 0);
#else
  *(u32x4*)l = *(const u32x4*)g;
#endif
}

__device__ __forceinline__ void wait_async() {
#if HAVE_ASYNC_LDS
#if __has_builtin(__builtin_amdgcn_s_wait_asynccnt)
  __builtin_amdgcn_s_wait_asynccnt(0);
#else
  asm volatile("s_wait_asynccnt 0x0" ::: "memory");
#endif
#endif
}

enum { ACT_NONE = 0, ACT_RELU = 1, ACT_QGELU = 2 };

// =====================================================================
// Weight prep: W (K x N, f32) -> WT (N x K, bf16), tiled 32x32 transpose.
// =====================================================================
__global__ __launch_bounds__(256)
void transpose_f32_to_bf16(const float* __restrict__ W, uint16_t* __restrict__ WT,
                           int K, int N)
{
  __shared__ float t[32][33];
  const int n0 = blockIdx.x * 32, k0 = blockIdx.y * 32;
  const int tx = threadIdx.x & 31, ty = threadIdx.x >> 5;   // 32 x 8
#pragma unroll
  for (int j = 0; j < 4; ++j)
    t[ty + 8 * j][tx] = W[(size_t)(k0 + ty + 8 * j) * N + n0 + tx];
  __syncthreads();
#pragma unroll
  for (int j = 0; j < 4; ++j)
    WT[(size_t)(n0 + ty + 8 * j) * K + k0 + tx] = f2bf(t[tx][ty + 8 * j]);
}

// =====================================================================
// bf16 WMMA GEMM: C(MxN) = act(A(MxK,bf16) * Bt(NxK,bf16)^T + bias) [+resid]
// 256 threads = 8 waves; tile 128x128, k-step 32, double-buffered LDS,
// async global->LDS staging pipelined against WMMA.
// =====================================================================
#define TM 128
#define TN 128
#define TK 32

__global__ __launch_bounds__(256)
void gemm_bf16_wmma(const uint16_t* __restrict__ A,    // M x K bf16
                    const uint16_t* __restrict__ Bt,   // N x K bf16 (pre-transposed)
                    const float*    __restrict__ bias, // N or nullptr
                    const float*    __restrict__ resid,// M x N f32 or nullptr
                    float*          __restrict__ outF, // f32 out (if non-null)
                    uint16_t*       __restrict__ outB, // bf16 out otherwise
                    int M, int N, int K, int act)
{
  __shared__ __align__(16) uint16_t As[2][TM][TK];   // 16 KB
  __shared__ __align__(16) uint16_t Bs[2][TN][TK];   // 16 KB  (Bs[n][k])

  const int tid  = threadIdx.x;
  const int lane = tid & 31;
  const int wave = tid >> 5;
  const int wm   = wave & 3;
  const int wn   = wave >> 2;
  const int mbase = blockIdx.y * TM;
  const int nbase = blockIdx.x * TN;

  f32x8 acc[2][4];
#pragma unroll
  for (int r = 0; r < 2; ++r)
#pragma unroll
    for (int c = 0; c < 4; ++c)
#pragma unroll
      for (int i = 0; i < 8; ++i) acc[r][c][i] = 0.0f;

  const int srow = tid >> 1;            // 0..127
  const int scol = (tid & 1) << 4;      // 0 or 16 halves

  auto stage = [&](int k0, int buf) {
    // ---- A tile (row-major copy; zero-fill ragged rows) ----
    uint16_t* la = &As[buf][srow][scol];
    const int grow = mbase + srow;
    if (grow < M) {
      const uint16_t* ga = A + (size_t)grow * K + k0 + scol;
      cp16(ga, la);
      cp16(ga + 8, la + 8);
    } else {
      u32x4 z; z[0] = z[1] = z[2] = z[3] = 0u;
      *(u32x4*)la = z; *(u32x4*)(la + 8) = z;
    }
    // ---- B tile (row-major copy of pre-transposed weights) ----
    const uint16_t* gb = Bt + (size_t)(nbase + srow) * K + k0 + scol;
    uint16_t* lb = &Bs[buf][srow][scol];
    cp16(gb, lb);
    cp16(gb + 8, lb + 8);
  };

  // fragment lane mapping (CDNA5 wave32 16-bit layouts)
  const int frow = lane & 15;
  const int akb  = (lane & 16) ? 8 : 0;
  const int bkb  = (lane & 16) ? 8 : 0;

  const int nk = K / TK;
  stage(0, 0);
  for (int t = 0; t < nk; ++t) {
    const int cur = t & 1;
    wait_async();
    __syncthreads();
    if (t + 1 < nk) stage((t + 1) * TK, cur ^ 1);

    FragAB afrag[2];
#pragma unroll
    for (int r = 0; r < 2; ++r) {
      const uint16_t* p = &As[cur][wm * 32 + r * 16 + frow][0];
      afrag[r].u[0] = *(const u32x4*)(p + akb);
      afrag[r].u[1] = *(const u32x4*)(p + akb + 16);
    }
    FragAB bfrag[4];
#pragma unroll
    for (int c = 0; c < 4; ++c) {
      const uint16_t* p = &Bs[cur][wn * 64 + c * 16 + frow][0];
      bfrag[c].u[0] = *(const u32x4*)(p + bkb);
      bfrag[c].u[1] = *(const u32x4*)(p + bkb + 16);
    }
#pragma unroll
    for (int r = 0; r < 2; ++r)
#pragma unroll
      for (int c = 0; c < 4; ++c)
        acc[r][c] = wmma_bf16(afrag[r], bfrag[c], acc[r][c]);
  }

  // ---- epilogue ----
#pragma unroll
  for (int r = 0; r < 2; ++r) {
#pragma unroll
    for (int c = 0; c < 4; ++c) {
      const int colg = nbase + wn * 64 + c * 16 + (lane & 15);
      const float bv = bias ? bias[colg] : 0.0f;
#pragma unroll
      for (int i = 0; i < 8; ++i) {
        const int rowg = mbase + wm * 32 + r * 16 + i + ((lane >> 4) << 3);
        if (rowg >= M) continue;
        float v = acc[r][c][i] + bv;
        if (act == ACT_RELU)       v = fmaxf(v, 0.0f);
        else if (act == ACT_QGELU) v = v / (1.0f + __expf(-1.702f * v));
        if (resid) v += resid[(size_t)rowg * N + colg];
        if (outF)  outF[(size_t)rowg * N + colg] = v;
        else       outB[(size_t)rowg * N + colg] = f2bf(v);
      }
    }
  }
}

// =====================================================================
// Fused per-(roi,head) attention (padded 64x64 tile, bf16 WMMA x2)
// =====================================================================
__global__ __launch_bounds__(128)
void attention_wmma(const uint16_t* __restrict__ Q,
                    const uint16_t* __restrict__ Km,
                    const uint16_t* __restrict__ V,
                    const float*    __restrict__ Mb,   // NROI x NPATCH
                    uint16_t*       __restrict__ O)
{
  __shared__ __align__(16) uint16_t qs[64][64];
  __shared__ __align__(16) uint16_t ks[64][64];
  __shared__ __align__(16) uint16_t vsT[64][64];  // vsT[d][m]
  __shared__ __align__(16) float    ss[64][64];
  __shared__ __align__(16) uint16_t ps[64][64];

  const int blk  = blockIdx.x;
  const int n    = blk / NHEAD;
  const int hh   = blk % NHEAD;
  const int tid  = threadIdx.x;
  const int lane = tid & 31;
  const int wave = tid >> 5;

  {
    const int r  = tid >> 1;
    const int c0 = (tid & 1) << 5;      // 0 or 32
    if (r < L_TOK) {
      const size_t g = ((size_t)(n * L_TOK + r)) * D_MODEL + hh * HDIM + c0;
#pragma unroll
      for (int i = 0; i < 4; ++i) {
        cp16(Q + g + i * 8, &qs[r][c0 + i * 8]);
        cp16(Km + g + i * 8, &ks[r][c0 + i * 8]);
      }
      const uint16_t* vp = V + g;
#pragma unroll
      for (int i = 0; i < 32; ++i) vsT[c0 + i][r] = vp[i];
    } else {
      u32x4 z; z[0] = z[1] = z[2] = z[3] = 0u;
#pragma unroll
      for (int i = 0; i < 4; ++i) {
        *(u32x4*)&qs[r][c0 + i * 8] = z;
        *(u32x4*)&ks[r][c0 + i * 8] = z;
      }
#pragma unroll
      for (int i = 0; i < 32; ++i) vsT[c0 + i][r] = 0;
    }
  }
  wait_async();
  __syncthreads();

  const int frow = lane & 15;
  const int akb  = (lane & 16) ? 8 : 0;
  const int bkb  = (lane & 16) ? 8 : 0;
  const int arow = wave * 16 + frow;

  // ---- scores = Q @ K^T ----
  f32x8 sc[4];
#pragma unroll
  for (int c = 0; c < 4; ++c)
#pragma unroll
    for (int i = 0; i < 8; ++i) sc[c][i] = 0.0f;
#pragma unroll
  for (int s = 0; s < 2; ++s) {
    FragAB a;
    a.u[0] = *(const u32x4*)&qs[arow][s * 32 + akb];
    a.u[1] = *(const u32x4*)&qs[arow][s * 32 + akb + 16];
#pragma unroll
    for (int c = 0; c < 4; ++c) {
      FragAB b;
      const int bm = c * 16 + frow;
      b.u[0] = *(const u32x4*)&ks[bm][s * 32 + bkb];
      b.u[1] = *(const u32x4*)&ks[bm][s * 32 + bkb + 16];
      sc[c] = wmma_bf16(a, b, sc[c]);
    }
  }
#pragma unroll
  for (int c = 0; c < 4; ++c)
#pragma unroll
    for (int i = 0; i < 8; ++i) {
      const int rr = wave * 16 + i + ((lane >> 4) << 3);
      const int cc = c * 16 + (lane & 15);
      ss[rr][cc] = sc[c][i] * 0.125f;
    }
  __syncthreads();

  // ---- diagonal overlap bias + softmax over first 50 cols ----
  if (tid < 64) {
    const int r = tid;
    if (r >= 1 && r < L_TOK) ss[r][r] += Mb[n * NPATCH + (r - 1)];
    float mx = -1e30f;
    for (int m = 0; m < L_TOK; ++m) mx = fmaxf(mx, ss[r][m]);
    float sum = 0.0f;
    for (int m = 0; m < L_TOK; ++m) { float e = __expf(ss[r][m] - mx); ss[r][m] = e; sum += e; }
    const float inv = 1.0f / sum;
    for (int m = 0; m < L_TOK; ++m) ps[r][m] = f2bf(ss[r][m] * inv);
    for (int m = L_TOK; m < 64; ++m) ps[r][m] = 0;
  }
  __syncthreads();

  // ---- O = P @ V ----
  f32x8 oc[4];
#pragma unroll
  for (int c = 0; c < 4; ++c)
#pragma unroll
    for (int i = 0; i < 8; ++i) oc[c][i] = 0.0f;
#pragma unroll
  for (int s = 0; s < 2; ++s) {
    FragAB a;
    a.u[0] = *(const u32x4*)&ps[arow][s * 32 + akb];
    a.u[1] = *(const u32x4*)&ps[arow][s * 32 + akb + 16];
#pragma unroll
    for (int c = 0; c < 4; ++c) {
      FragAB b;
      const int bd = c * 16 + frow;
      b.u[0] = *(const u32x4*)&vsT[bd][s * 32 + bkb];
      b.u[1] = *(const u32x4*)&vsT[bd][s * 32 + bkb + 16];
      oc[c] = wmma_bf16(a, b, oc[c]);
    }
  }
#pragma unroll
  for (int c = 0; c < 4; ++c)
#pragma unroll
    for (int i = 0; i < 8; ++i) {
      const int rr = wave * 16 + i + ((lane >> 4) << 3);
      const int cc = c * 16 + (lane & 15);
      if (rr < L_TOK)
        O[((size_t)(n * L_TOK + rr)) * D_MODEL + hh * HDIM + cc] = f2bf(oc[c][i]);
    }
}

// =====================================================================
// LayerNorm (f32 in) -> bf16 out.  One block per row, 256 threads, D=768.
// =====================================================================
__global__ __launch_bounds__(256)
void layernorm_bf16(const float* __restrict__ x, const float* __restrict__ s,
                    const float* __restrict__ b, uint16_t* __restrict__ h)
{
  __shared__ float red[256];
  const int row = blockIdx.x;
  const int tid = threadIdx.x;
  const float* xr = x + (size_t)row * D_MODEL;
  const float v0 = xr[tid], v1 = xr[tid + 256], v2 = xr[tid + 512];

  red[tid] = v0 + v1 + v2;
  __syncthreads();
  for (int k = 128; k > 0; k >>= 1) { if (tid < k) red[tid] += red[tid + k]; __syncthreads(); }
  const float mean = red[0] * (1.0f / 768.0f);
  __syncthreads();

  const float d0 = v0 - mean, d1 = v1 - mean, d2 = v2 - mean;
  red[tid] = d0 * d0 + d1 * d1 + d2 * d2;
  __syncthreads();
  for (int k = 128; k > 0; k >>= 1) { if (tid < k) red[tid] += red[tid + k]; __syncthreads(); }
  const float rstd = rsqrtf(red[0] * (1.0f / 768.0f) + 1e-5f);

  uint16_t* hr = h + (size_t)row * D_MODEL;
  hr[tid]       = f2bf(d0 * rstd * s[tid]       + b[tid]);
  hr[tid + 256] = f2bf(d1 * rstd * s[tid + 256] + b[tid + 256]);
  hr[tid + 512] = f2bf(d2 * rstd * s[tid + 512] + b[tid + 512]);
}

// ---------------- small helper kernels ----------------
__global__ void patch_gather(const float* __restrict__ frame, uint16_t* __restrict__ patches)
{
  const int idx = blockIdx.x * 256 + threadIdx.x;
  if (idx >= NPATCH * 3072) return;
  const int p = idx / 3072, e = idx % 3072;
  const int c = e >> 10, rem = e & 1023, py = rem >> 5, px = rem & 31;
  const int gr = p / 7, gc = p % 7;
  patches[idx] = f2bf(frame[c * 224 * 224 + (gr * 32 + py) * 224 + gc * 32 + px]);
}

__global__ void overlap_kernel(const float* __restrict__ bboxes, float* __restrict__ Mout)
{
  const int n = blockIdx.x, p = threadIdx.x;
  if (p >= NPATCH) return;
  const float bx1 = bboxes[n*4+0], by1 = bboxes[n*4+1], bx2 = bboxes[n*4+2], by2 = bboxes[n*4+3];
  const int pr = p / 7, pc = p % 7;
  const float px1 = pc * 32.0f, py1 = pr * 32.0f;
  float iw = fminf(bx2, px1 + 32.0f) - fmaxf(bx1, px1); iw = fmaxf(iw, 0.0f);
  float ih = fminf(by2, py1 + 32.0f) - fmaxf(by1, py1); ih = fmaxf(ih, 0.0f);
  Mout[n * NPATCH + p] = iw * ih * (1.0f / 1024.0f);
}

__global__ __launch_bounds__(256)
void build_x(const float* __restrict__ feat, const float* __restrict__ cls,
             float* __restrict__ x)
{
  const int row = blockIdx.x;
  const int l = row % L_TOK;
  const float* src = (l == 0) ? cls : (feat + (size_t)(l - 1) * D_MODEL);
  for (int j = threadIdx.x; j < D_MODEL; j += 256)
    x[(size_t)row * D_MODEL + j] = src[j];
}

__global__ __launch_bounds__(256)
void extract_cls(const float* __restrict__ x, float* __restrict__ out)
{
  const int n = blockIdx.x;
  for (int j = threadIdx.x; j < D_MODEL; j += 256)
    out[(size_t)n * D_MODEL + j] = x[(size_t)(n * L_TOK) * D_MODEL + j];
}

// =====================================================================
// host launcher
// =====================================================================
extern "C" void kernel_launch(void* const* d_in, const int* in_sizes, int n_in,
                              void* d_out, int out_size, void* d_ws, size_t ws_size,
                              hipStream_t stream)
{
  const float* frame   = (const float*)d_in[0];
  const float* bboxes  = (const float*)d_in[1];
  const float* conv_w  = (const float*)d_in[2];
  const float* cls_emb = (const float*)d_in[3];
  const float* ln1_s   = (const float*)d_in[4];
  const float* ln1_b   = (const float*)d_in[5];
  const float* wq      = (const float*)d_in[6];
  const float* bq      = (const float*)d_in[7];
  const float* wk      = (const float*)d_in[8];
  const float* bk      = (const float*)d_in[9];
  const float* wv      = (const float*)d_in[10];
  const float* bv      = (const float*)d_in[11];
  const float* wo      = (const float*)d_in[12];
  const float* bo      = (const float*)d_in[13];
  const float* ln2_s   = (const float*)d_in[14];
  const float* ln2_b   = (const float*)d_in[15];
  const float* w1      = (const float*)d_in[16];
  const float* b1      = (const float*)d_in[17];
  const float* w2      = (const float*)d_in[18];
  const float* b2      = (const float*)d_in[19];

  char* ws = (char*)d_ws;
  size_t off = 0;
  auto walloc = [&](size_t bytes) -> void* {
    void* p = ws + off;
    off = (off + bytes + 255) & ~(size_t)255;
    return p;
  };
  float*    x       = (float*)   walloc((size_t)ROWS * D_MODEL * 4);
  uint16_t* h       = (uint16_t*)walloc((size_t)ROWS * D_MODEL * 2);
  uint16_t* qb      = (uint16_t*)walloc((size_t)ROWS * D_MODEL * 2);
  uint16_t* kb2     = (uint16_t*)walloc((size_t)ROWS * D_MODEL * 2);
  uint16_t* vb2     = (uint16_t*)walloc((size_t)ROWS * D_MODEL * 2);
  uint16_t* ob      = (uint16_t*)walloc((size_t)ROWS * D_MODEL * 2);
  uint16_t* mid     = (uint16_t*)walloc((size_t)ROWS * MLPD * 2);
  uint16_t* patches = (uint16_t*)walloc((size_t)NPATCH * 3072 * 2);
  float*    feat    = (float*)   walloc((size_t)NPATCH * D_MODEL * 4);
  float*    Mbuf    = (float*)   walloc((size_t)NROI * NPATCH * 4);
  // per-layer bf16 transposed weights (N x K)
  uint16_t* convT   = (uint16_t*)walloc((size_t)D_MODEL * 3072 * 2);
  uint16_t* qT      = (uint16_t*)walloc((size_t)D_MODEL * D_MODEL * 2);
  uint16_t* kT      = (uint16_t*)walloc((size_t)D_MODEL * D_MODEL * 2);
  uint16_t* vT      = (uint16_t*)walloc((size_t)D_MODEL * D_MODEL * 2);
  uint16_t* oT      = (uint16_t*)walloc((size_t)D_MODEL * D_MODEL * 2);
  uint16_t* w1T     = (uint16_t*)walloc((size_t)MLPD * D_MODEL * 2);
  uint16_t* w2T     = (uint16_t*)walloc((size_t)D_MODEL * MLPD * 2);
  (void)ws_size; (void)in_sizes; (void)n_in; (void)out_size;

  // ---- stem ----
  patch_gather<<<(NPATCH * 3072 + 255) / 256, 256, 0, stream>>>(frame, patches);
  transpose_f32_to_bf16<<<dim3(D_MODEL / 32, 3072 / 32), 256, 0, stream>>>(
      conv_w, convT, 3072, D_MODEL);
  gemm_bf16_wmma<<<dim3(D_MODEL / TN, 1), 256, 0, stream>>>(
      patches, convT, nullptr, nullptr, feat, nullptr,
      NPATCH, D_MODEL, 3072, ACT_RELU);
  overlap_kernel<<<NROI, 64, 0, stream>>>(bboxes, Mbuf);
  build_x<<<ROWS, 256, 0, stream>>>(feat, cls_emb, x);

  const dim3 gD(D_MODEL / TN, ROWS / TM);   // (6, 25)
  const dim3 gM(MLPD / TN,    ROWS / TM);   // (24, 25)
  const dim3 tDD(D_MODEL / 32, D_MODEL / 32);
  const dim3 tDM(MLPD / 32,    D_MODEL / 32);   // for W1 (768 x 3072)
  const dim3 tMD(D_MODEL / 32, MLPD / 32);      // for W2 (3072 x 768)

  for (int l = 0; l < NLAYER; ++l) {
    const size_t wofs = (size_t)l * D_MODEL * D_MODEL;
    const size_t mofs = (size_t)l * D_MODEL * MLPD;

    // weight prep for this layer (bandwidth noise: ~42 MB @ 23.3 TB/s)
    transpose_f32_to_bf16<<<tDD, 256, 0, stream>>>(wq + wofs, qT, D_MODEL, D_MODEL);
    transpose_f32_to_bf16<<<tDD, 256, 0, stream>>>(wk + wofs, kT, D_MODEL, D_MODEL);
    transpose_f32_to_bf16<<<tDD, 256, 0, stream>>>(wv + wofs, vT, D_MODEL, D_MODEL);
    transpose_f32_to_bf16<<<tDD, 256, 0, stream>>>(wo + wofs, oT, D_MODEL, D_MODEL);
    transpose_f32_to_bf16<<<tDM, 256, 0, stream>>>(w1 + mofs, w1T, D_MODEL, MLPD);
    transpose_f32_to_bf16<<<tMD, 256, 0, stream>>>(w2 + mofs, w2T, MLPD, D_MODEL);

    layernorm_bf16<<<ROWS, 256, 0, stream>>>(x, ln1_s + l * D_MODEL, ln1_b + l * D_MODEL, h);

    gemm_bf16_wmma<<<gD, 256, 0, stream>>>(h, qT, bq + l * D_MODEL,
        nullptr, nullptr, qb, ROWS, D_MODEL, D_MODEL, ACT_NONE);
    gemm_bf16_wmma<<<gD, 256, 0, stream>>>(h, kT, bk + l * D_MODEL,
        nullptr, nullptr, kb2, ROWS, D_MODEL, D_MODEL, ACT_NONE);
    gemm_bf16_wmma<<<gD, 256, 0, stream>>>(h, vT, bv + l * D_MODEL,
        nullptr, nullptr, vb2, ROWS, D_MODEL, D_MODEL, ACT_NONE);

    attention_wmma<<<NROI * NHEAD, 128, 0, stream>>>(qb, kb2, vb2, Mbuf, ob);

    gemm_bf16_wmma<<<gD, 256, 0, stream>>>(ob, oT, bo + l * D_MODEL,
        x, x, nullptr, ROWS, D_MODEL, D_MODEL, ACT_NONE);

    layernorm_bf16<<<ROWS, 256, 0, stream>>>(x, ln2_s + l * D_MODEL, ln2_b + l * D_MODEL, h);

    gemm_bf16_wmma<<<gM, 256, 0, stream>>>(h, w1T, b1 + (size_t)l * MLPD,
        nullptr, nullptr, mid, ROWS, MLPD, D_MODEL, ACT_QGELU);
    gemm_bf16_wmma<<<gD, 256, 0, stream>>>(mid, w2T, b2 + l * D_MODEL,
        x, x, nullptr, ROWS, D_MODEL, MLPD, ACT_NONE);
  }

  extract_cls<<<NROI, 256, 0, stream>>>(x, (float*)d_out);
}
